// FeedForwardQuantum_5368709120606
// MI455X (gfx1250) — matrix operand order repaired and checked by reading, not compiled, not run on hardware
//
#include <hip/hip_runtime.h>

typedef __attribute__((ext_vector_type(16))) _Float16 v16h;
typedef __attribute__((ext_vector_type(8)))  _Float16 v8h;
typedef __attribute__((ext_vector_type(8)))  float    v8f;
typedef __attribute__((ext_vector_type(2)))  float    f32x2;
typedef __attribute__((ext_vector_type(4)))  float    f32x4;

constexpr int kNQ = 8;
constexpr int kE  = 512;
constexpr int kS  = 2048;
constexpr int kB  = 16;
constexpr int kN  = kB * kS;   // 32768 rows

// ---------------------------------------------------------------------------
// Kernel 1: h = x @ w_in^T + b_in  (WMMA f16, f32 accum), then the analytic
// quantum collapse: meas[w] = prefix products of cos(theta_w). Writes meas as
// f16 [kN][8] into workspace (row n = b*S + s).
//
// Note: C columns 8..15 are never read back, so the B operand's columns 8..15
// may hold garbage -> load w_in row (m & 7) unconditionally (no EXEC branch
// in the hot loop).
// ---------------------------------------------------------------------------
__global__ __launch_bounds__(256)
void k_encode(const float* __restrict__ x,
              const float* __restrict__ w_in,
              const float* __restrict__ b_in,
              const float* __restrict__ rx_theta,
              _Float16* __restrict__ meas)
{
    __shared__ float hbuf[8][16][16];   // per-wave C tiles
    const int wave = threadIdx.x >> 5;
    const int lane = threadIdx.x & 31;
    const int m    = lane & 15;
    const int hi   = lane >> 4;
    const int tile = blockIdx.x * 8 + wave;
    const int n0   = tile * 16;

    const float* xrow = x + (size_t)(n0 + m) * kE;               // streamed once
    const float* wrow = w_in + (size_t)(m & 7) * kE + 16 * hi;   // tiny, cached

    v8f acc = {};
    #pragma unroll 2
    for (int c = 0; c < kE / 32; ++c) {
        const int k0 = c * 32;
        // A operand: 16x32 f16, contiguous K layout:
        // VGPR v holds K pair (K,K+1), K = 2v + (hi?16:0)
        v16h a;
        #pragma unroll
        for (int v = 0; v < 8; ++v) {
            const int K = 2 * v + (hi ? 16 : 0);
            const f32x2 f =
                __builtin_nontemporal_load((const f32x2*)(xrow + k0 + K));
            a[2 * v]     = (_Float16)f.x;
            a[2 * v + 1] = (_Float16)f.y;
        }
        // B operand: 32x16 f16; lane column q = m, halves j -> K = j + 16*hi
        v16h b;
        #pragma unroll
        for (int j4 = 0; j4 < 4; ++j4) {
            const f32x4 f = *(const f32x4*)(wrow + k0 + 4 * j4);
            b[4 * j4 + 0] = (_Float16)f.x;
            b[4 * j4 + 1] = (_Float16)f.y;
            b[4 * j4 + 2] = (_Float16)f.z;
            b[4 * j4 + 3] = (_Float16)f.w;
        }
        acc = __builtin_amdgcn_wmma_f32_16x16x32_f16(false, a, false, b,
                                                     (short)0, acc, false, false);
    }

    // Scatter C tile to LDS: lane column q = m, row = r + 8*hi
    #pragma unroll
    for (int r = 0; r < 8; ++r)
        hbuf[wave][r + 8 * hi][m] = acc[r];
    __syncthreads();

    // Epilogue: one row per lane (lanes 0-15). Analytic quantum layer.
    if (lane < 16) {
        const int row = lane;
        float h[kNQ];
        #pragma unroll
        for (int q = 0; q < kNQ; ++q) h[q] = hbuf[wave][row][q] + b_in[q];

        float th[kNQ];
        th[0] = h[0] + rx_theta[0];
        th[1] = h[1] + rx_theta[1];
        th[2] = h[2] + h[3] + rx_theta[2];   // ENC_WIRES routes angle 3 -> wire 2
        th[3] = rx_theta[3];                 // wire 3: trainable only
        th[4] = h[4] + rx_theta[4];
        th[5] = h[5] + rx_theta[5];
        th[6] = h[6] + rx_theta[6];
        th[7] = h[7] + rx_theta[7];

        float cw[kNQ];
        #pragma unroll
        for (int q = 0; q < kNQ; ++q) cw[q] = cosf(th[q]);

        // <Z_w> = prod_{0..w} cos  (w>=1);  <Z_0> = prod_{1..7} cos
        float mz[kNQ];
        float pre = cw[0];
        #pragma unroll
        for (int w = 1; w < kNQ; ++w) { pre *= cw[w]; mz[w] = pre; }
        float p0 = cw[1];
        #pragma unroll
        for (int w = 2; w < kNQ; ++w) p0 *= cw[w];
        mz[0] = p0;

        v8h o;
        #pragma unroll
        for (int q = 0; q < kNQ; ++q) o[q] = (_Float16)mz[q];
        *(v8h*)(meas + (size_t)(n0 + row) * kNQ) = o;
    }
}

// ---------------------------------------------------------------------------
// Kernel 2: out[s][b][e] = sum_q meas[b*S+s][q] * w_out[e][q] + b_out[e].
// One WMMA (K=8 zero-padded to 32) per 16x16 output tile: rows = batch b,
// cols = 16 consecutive e. A's K>=8 slots hit zero B rows, so only B needs
// explicit zeros for K>=8 (hi lanes / j>=8).
// ---------------------------------------------------------------------------
__global__ __launch_bounds__(256)
void k_decode(const _Float16* __restrict__ meas,
              const float* __restrict__ w_out,
              const float* __restrict__ b_out,
              float* __restrict__ out)
{
    const int wave = threadIdx.x >> 5;
    const int lane = threadIdx.x & 31;
    const int m    = lane & 15;
    const int hi   = lane >> 4;
    const int t    = blockIdx.x * 8 + wave;
    const int s    = t >> 5;            // 0..2047
    const int e0   = (t & 31) << 4;     // 0..511 step 16

    // A: row m = batch index b; a[0..7] = meas[m*S+s][0..7] (K=0..7).
    // K>=8 slots (a[8..15] and all of hi lanes) hit zero B rows -> don't care.
    v16h a;
    {
        const v8h lo = *(const v8h*)(meas + ((size_t)m * kS + s) * kNQ);
        #pragma unroll
        for (int j = 0; j < 8; ++j) { a[j] = lo[j]; a[8 + j] = lo[j]; }
    }
    // B: column n = m -> e = e0+m; b[j] = w_out[e][j] for K=j<8 (lanes 0-15);
    // K>=8 (j>=8 on lo lanes, all of hi lanes) MUST be zero.
    v16h b{};
    if (!hi) {
        const float* wr = w_out + (size_t)(e0 + m) * kNQ;
        const f32x4 f0 = *(const f32x4*)(wr);
        const f32x4 f1 = *(const f32x4*)(wr + 4);
        b[0] = (_Float16)f0.x; b[1] = (_Float16)f0.y;
        b[2] = (_Float16)f0.z; b[3] = (_Float16)f0.w;
        b[4] = (_Float16)f1.x; b[5] = (_Float16)f1.y;
        b[6] = (_Float16)f1.z; b[7] = (_Float16)f1.w;
    }

    v8f c = {};
    c = __builtin_amdgcn_wmma_f32_16x16x32_f16(false, a, false, b,
                                               (short)0, c, false, false);

    const float bias = b_out[e0 + m];
    float* ocol = out + (size_t)(s * 16) * kE + e0 + m;  // lane's column
    #pragma unroll
    for (int r = 0; r < 8; ++r)
        __builtin_nontemporal_store(c[r] + bias,
                                    ocol + (size_t)(r + 8 * hi) * kE);
}

// ---------------------------------------------------------------------------
extern "C" void kernel_launch(void* const* d_in, const int* in_sizes, int n_in,
                              void* d_out, int out_size, void* d_ws, size_t ws_size,
                              hipStream_t stream)
{
    const float* x        = (const float*)d_in[0];
    const float* w_in     = (const float*)d_in[1];
    const float* b_in     = (const float*)d_in[2];
    const float* rx_theta = (const float*)d_in[3];
    const float* w_out    = (const float*)d_in[4];
    const float* b_out    = (const float*)d_in[5];
    float* out            = (float*)d_out;
    _Float16* meas        = (_Float16*)d_ws;   // 32768 * 8 f16 = 512 KB

    // GEMM1 + quantum collapse: 2048 tiles / 8 waves per block
    k_encode<<<kN / 16 / 8, 256, 0, stream>>>(x, w_in, b_in, rx_theta, meas);
    // GEMM2: 2048*32 tiles / 8 waves per block
    k_decode<<<(kS * (kE / 16)) / 8, 256, 0, stream>>>(meas, w_out, b_out, out);
}
